// LiteMLA_90967407330011
// MI455X (gfx1250) — matrix operand order repaired and checked by reading, not compiled, not run on hardware
//
#include <hip/hip_runtime.h>
#include <hip/hip_bf16.h>

// ---------------------------------------------------------------------------
// LiteMLA block for MI455X (gfx1250, wave32, WMMA).
// Pipeline:
//   P0: convert w_qkv / w_proj to f16; fold BN into alpha/bias
//   P1: transpose+convert x (B,256,HW) f32 -> xT (B,HW,256) f16
//   P2: qkv GEMM   [768x256]x[256x4096] per batch via v_wmma_f32_16x16x32_f16
//   P3: fused depthwise 5x5 + grouped 8x8 pointwise -> agg (f16)
//   P4: kv = relu(K) [Vt|1] per (b,head) via WMMA, 4-wave split-K + LDS reduce
//   P5: out = relu(Q)*kv, normalize, write attT (B,HW,512) f16
//   P6: proj GEMM  [256x512]x[512x4096] via WMMA + fused BN affine -> d_out f32
// GEMM inner loops are software-pipelined (double-buffered fragments) so each
// v_wmma can issue with outstanding loads instead of s_wait_loadcnt 0.
// ---------------------------------------------------------------------------

typedef __attribute__((ext_vector_type(16))) _Float16 v16h;
typedef __attribute__((ext_vector_type(8)))  _Float16 h8;
typedef __attribute__((ext_vector_type(8)))  float    v8f;

#define NB    8
#define CIN   256
#define COUT  256
#define CQKV  768
#define CATT  512
#define HWN   4096
#define WID   64
#define NG    96
#define NHEAD 64

static __device__ inline v8f wmma_f16(v16h a, v16h b, v8f c) {
    return __builtin_amdgcn_wmma_f32_16x16x32_f16(
        /*neg_a=*/false, a, /*neg_b=*/false, b,
        /*c_mod=*/(short)0, c, /*reuse_a=*/false, /*reuse_b=*/false);
}

// A fragment (16x32 f16): lane holds 8 halves at p, 8 halves at p+16.
static __device__ inline v16h frag_from2(const _Float16* p0, const _Float16* p1) {
    h8 x = *(const h8*)p0;
    h8 y = *(const h8*)p1;
    v16h r;
#pragma unroll
    for (int i = 0; i < 8; ++i) { r[i] = x[i]; r[8 + i] = y[i]; }
    return r;
}
// B fragment (32x16 f16): lane holds 8 halves at p, 8 halves at p+16.
static __device__ inline v16h frag_b16(const _Float16* p) {
    return frag_from2(p, p + 16);
}
static __device__ inline v16h relu16(v16h a) {
    v16h r;
#pragma unroll
    for (int i = 0; i < 16; ++i) {
        _Float16 v = a[i];
        r[i] = (v > (_Float16)0.0f) ? v : (_Float16)0.0f;
    }
    return r;
}
static __device__ inline v16h fill16(_Float16 v) {
    v16h r;
#pragma unroll
    for (int i = 0; i < 16; ++i) r[i] = v;
    return r;
}

// ---------------- P0: small conversions -----------------------------------
__global__ void k_cvt_f16(const float* __restrict__ in, _Float16* __restrict__ out, int n) {
    int i = blockIdx.x * blockDim.x + threadIdx.x;
    if (i < n) out[i] = (_Float16)in[i];
}

__global__ void k_bn_fold(const float* __restrict__ gamma, const float* __restrict__ beta,
                          const float* __restrict__ mean,  const float* __restrict__ var,
                          float* __restrict__ alpha, float* __restrict__ bias) {
    int i = threadIdx.x;
    if (i < COUT) {
        float a = gamma[i] * rsqrtf(var[i] + 1e-5f);
        alpha[i] = a;
        bias[i]  = beta[i] - mean[i] * a;
    }
}

// ---------------- P1: x (B,C,HW) f32 -> xT (B,HW,C) f16 --------------------
__global__ void k_xpose(const float* __restrict__ x, _Float16* __restrict__ xT) {
    __shared__ float t[32][33];
    const int b  = blockIdx.z;
    const int p0 = blockIdx.x * 32;
    const int c0 = blockIdx.y * 32;
    const int tx = threadIdx.x, ty = threadIdx.y;
    const float* xb = x + (size_t)b * CIN * HWN;
#pragma unroll
    for (int i = 0; i < 4; ++i) {
        int c = c0 + ty + i * 8;
        t[ty + i * 8][tx] = xb[(size_t)c * HWN + p0 + tx];
    }
    __syncthreads();
    _Float16* ob = xT + (size_t)b * HWN * CIN;
#pragma unroll
    for (int i = 0; i < 4; ++i) {
        int p = p0 + ty + i * 8;
        ob[(size_t)p * CIN + c0 + tx] = (_Float16)t[tx][ty + i * 8];
    }
}

// ---------------- software-pipelined 16x64 WMMA tile core ------------------
// A: [M x K] row-major f16 (row stride = K), B: [N x K] row-major f16
// (row stride = K, i.e. column n of the logical KxN matrix is contiguous).
template <int K>
static __device__ inline void gemm_tile_16x64(const _Float16* __restrict__ arow,
                                              const _Float16* __restrict__ bbase,
                                              int lo, int hi, v8f acc[4]) {
    // bbase points at row (n0 + lo) of B; the 4 N-tiles are 16 rows apart.
    const _Float16* ap = arow + hi * 8;
    const _Float16* bp = bbase + hi * 8;
    v16h a_cur = frag_from2(ap, ap + 16);
    v16h b_cur[4];
#pragma unroll
    for (int t = 0; t < 4; ++t) b_cur[t] = frag_b16(bp + (size_t)t * 16 * K);

#pragma unroll
    for (int k = 0; k < K; k += 32) {
        v16h a_nxt, b_nxt[4];
        if (k + 32 < K) {                       // resolved at compile time
            const _Float16* an = ap + k + 32;
            a_nxt = frag_from2(an, an + 16);
#pragma unroll
            for (int t = 0; t < 4; ++t)
                b_nxt[t] = frag_b16(bp + (size_t)t * 16 * K + k + 32);
        }
#pragma unroll
        for (int t = 0; t < 4; ++t) acc[t] = wmma_f16(a_cur, b_cur[t], acc[t]);
        if (k + 32 < K) {
            a_cur = a_nxt;
#pragma unroll
            for (int t = 0; t < 4; ++t) b_cur[t] = b_nxt[t];
        }
    }
}

// ---------------- P2: qkv GEMM (WMMA) --------------------------------------
__global__ void k_gemm_qkv(const _Float16* __restrict__ wq,
                           const _Float16* __restrict__ xT,
                           _Float16* __restrict__ qkv) {
    const int lane = threadIdx.x;
    const int lo = lane & 15, hi = lane >> 4;
    const int b  = blockIdx.z;
    const int m0 = blockIdx.y * 16;
    const int n0 = blockIdx.x * 64;

    v8f acc[4] = {v8f{}, v8f{}, v8f{}, v8f{}};
    const _Float16* arow  = wq + (size_t)(m0 + lo) * CIN;
    const _Float16* bbase = xT + (size_t)b * HWN * CIN + (size_t)(n0 + lo) * CIN;
    gemm_tile_16x64<CIN>(arow, bbase, lo, hi, acc);

    _Float16* ob = qkv + (size_t)b * CQKV * HWN;
#pragma unroll
    for (int j = 0; j < 8; ++j) {
        int row = m0 + j + hi * 8;
#pragma unroll
        for (int t = 0; t < 4; ++t)
            ob[(size_t)row * HWN + n0 + t * 16 + lo] = (_Float16)acc[t][j];
    }
}

// ---------------- P3: fused depthwise 5x5 + grouped 8x8 pointwise ----------
__global__ void k_dwpw(const _Float16* __restrict__ qkv,
                       const float* __restrict__ wdw,
                       const float* __restrict__ wpw,
                       _Float16* __restrict__ agg) {
    __shared__ _Float16 tile[8][20][20];
    __shared__ float dwW[200];
    __shared__ float pwW[64];
    const int bg = blockIdx.z;
    const int b = bg / NG, g = bg % NG;
    const int tx = threadIdx.x, ty = threadIdx.y;
    const int tid = ty * 16 + tx;
    const int x0 = blockIdx.x * 16 - 2;
    const int y0 = blockIdx.y * 16 - 2;
    const _Float16* qb = qkv + (size_t)b * CQKV * HWN + (size_t)(g * 8) * HWN;

    for (int idx = tid; idx < 3200; idx += 256) {
        int c = idx / 400, rem = idx % 400;
        int r = rem / 20, cc = rem % 20;
        int gy = y0 + r, gx = x0 + cc;
        _Float16 v = (_Float16)0.0f;
        if (gy >= 0 && gy < WID && gx >= 0 && gx < WID)
            v = qb[(size_t)c * HWN + gy * WID + gx];
        tile[c][r][cc] = v;
    }
    if (tid < 200) dwW[tid] = wdw[(size_t)(g * 8) * 25 + tid];
    if (tid < 64)  pwW[tid] = wpw[(size_t)g * 64 + tid];
    __syncthreads();

    float dwv[8];
#pragma unroll
    for (int c = 0; c < 8; ++c) {
        float s = 0.f;
#pragma unroll
        for (int i = 0; i < 5; ++i)
#pragma unroll
            for (int j = 0; j < 5; ++j)
                s += dwW[c * 25 + i * 5 + j] * (float)tile[c][ty + i][tx + j];
        dwv[c] = s;
    }
    const int p = (blockIdx.y * 16 + ty) * WID + blockIdx.x * 16 + tx;
    _Float16* ab = agg + (size_t)b * CQKV * HWN;
#pragma unroll
    for (int o = 0; o < 8; ++o) {
        float s = 0.f;
#pragma unroll
        for (int i = 0; i < 8; ++i) s += pwW[o * 8 + i] * dwv[i];
        ab[(size_t)(g * 8 + o) * HWN + p] = (_Float16)s;
    }
}

// ---------------- P4: kv = relu(K)(8x4096) x [V|1](4096x9) via WMMA --------
// Block = 4 waves, each wave reduces 1024 of the 4096 spatial positions.
__global__ void k_attn_kv(const _Float16* __restrict__ qkv,
                          const _Float16* __restrict__ agg,
                          float* __restrict__ kvout) {
    const int bh = blockIdx.x;
    const int b = bh >> 6, h = bh & 63;
    const int wv = threadIdx.x >> 5;
    const int lane = threadIdx.x & 31;
    const int lo = lane & 15, hi = lane >> 4;

    const _Float16* ms = (h < 32)
        ? qkv + (size_t)b * CQKV * HWN + (size_t)(h * 24) * HWN
        : agg + (size_t)b * CQKV * HWN + (size_t)((h - 32) * 24) * HWN;
    const _Float16* kch = ms + (size_t)8 * HWN;
    const _Float16* vch = ms + (size_t)16 * HWN;

    const bool isK = (lo < 8);
    const _Float16* kp = kch + (size_t)(lo & 7) * HWN + hi * 8;   // A chunks
    const _Float16* vp = vch + (size_t)(lo & 7) * HWN + hi * 8;   // B chunks
    const v16h bfix = (lo == 8) ? fill16((_Float16)1.0f) : fill16((_Float16)0.0f);
    const v16h zero = fill16((_Float16)0.0f);

    v8f acc{};
    const int pbeg = wv * 1024;
    v16h a_cur = isK ? relu16(frag_from2(kp + pbeg, kp + pbeg + 16)) : zero;
    v16h b_cur = isK ? frag_b16(vp + pbeg) : bfix;
#pragma unroll 4
    for (int pb = pbeg; pb < pbeg + 1024; pb += 32) {
        v16h a_nxt = zero, b_nxt = bfix;
        if (pb + 32 < pbeg + 1024) {
            if (isK) {
                a_nxt = relu16(frag_from2(kp + pb + 32, kp + pb + 48));
                b_nxt = frag_b16(vp + pb + 32);
            }
        }
        acc = wmma_f16(a_cur, b_cur, acc);
        a_cur = a_nxt;
        b_cur = b_nxt;
    }

    __shared__ float red[4][72];
    if (hi == 0 && lo < 9) {
#pragma unroll
        for (int j = 0; j < 8; ++j) red[wv][j * 9 + lo] = acc[j]; // (d=j, e=lo)
    }
    __syncthreads();
    if (threadIdx.x < 72) {
        float s = red[0][threadIdx.x] + red[1][threadIdx.x] +
                  red[2][threadIdx.x] + red[3][threadIdx.x];
        kvout[(size_t)bh * 72 + threadIdx.x] = s;
    }
}

// ---------------- P5: out = relu(Q)*kv, normalize -> attT (B,HW,512) f16 ---
__global__ void k_attn_out(const _Float16* __restrict__ qkv,
                           const _Float16* __restrict__ agg,
                           const float* __restrict__ kv,
                           _Float16* __restrict__ attT) {
    __shared__ float kvs[72];
    const int b = blockIdx.z, h = blockIdx.y;
    const int tid = threadIdx.x;
    if (tid < 72) kvs[tid] = kv[((size_t)b * 64 + h) * 72 + tid];
    __syncthreads();

    const _Float16* ms = (h < 32)
        ? qkv + (size_t)b * CQKV * HWN + (size_t)(h * 24) * HWN
        : agg + (size_t)b * CQKV * HWN + (size_t)((h - 32) * 24) * HWN;
    const int p = blockIdx.x * 256 + tid;

    float out[9];
#pragma unroll
    for (int e = 0; e < 9; ++e) out[e] = 0.f;
#pragma unroll
    for (int d = 0; d < 8; ++d) {
        float q = (float)ms[(size_t)d * HWN + p];
        q = fmaxf(q, 0.f);
#pragma unroll
        for (int e = 0; e < 9; ++e) out[e] += q * kvs[d * 9 + e];
    }
    float r = 1.0f / (out[8] + 1e-15f);
    h8 res;
#pragma unroll
    for (int d = 0; d < 8; ++d) res[d] = (_Float16)(out[d] * r);
    *(h8*)(attT + ((size_t)b * HWN + p) * CATT + h * 8) = res;
}

// ---------------- P6: proj GEMM (WMMA) + BN affine -> d_out f32 ------------
__global__ void k_gemm_proj(const _Float16* __restrict__ wp,
                            const _Float16* __restrict__ attT,
                            const float* __restrict__ alpha,
                            const float* __restrict__ bias,
                            float* __restrict__ out) {
    const int lane = threadIdx.x;
    const int lo = lane & 15, hi = lane >> 4;
    const int b  = blockIdx.z;
    const int m0 = blockIdx.y * 16;
    const int n0 = blockIdx.x * 64;

    v8f acc[4] = {v8f{}, v8f{}, v8f{}, v8f{}};
    const _Float16* arow  = wp + (size_t)(m0 + lo) * CATT;
    const _Float16* bbase = attT + (size_t)b * HWN * CATT + (size_t)(n0 + lo) * CATT;
    gemm_tile_16x64<CATT>(arow, bbase, lo, hi, acc);

    float* ob = out + (size_t)b * COUT * HWN;
#pragma unroll
    for (int j = 0; j < 8; ++j) {
        int row = m0 + j + hi * 8;
        float al = alpha[row], bi = bias[row];
#pragma unroll
        for (int t = 0; t < 4; ++t)
            ob[(size_t)row * HWN + n0 + t * 16 + lo] = acc[t][j] * al + bi;
    }
}

// ---------------------------------------------------------------------------
extern "C" void kernel_launch(void* const* d_in, const int* in_sizes, int n_in,
                              void* d_out, int out_size, void* d_ws, size_t ws_size,
                              hipStream_t stream) {
    const float* x      = (const float*)d_in[0];
    const float* w_qkv  = (const float*)d_in[1];
    const float* w_dw   = (const float*)d_in[2];
    const float* w_pw   = (const float*)d_in[3];
    const float* w_proj = (const float*)d_in[4];
    const float* gamma  = (const float*)d_in[5];
    const float* beta   = (const float*)d_in[6];
    const float* mean   = (const float*)d_in[7];
    const float* var    = (const float*)d_in[8];
    float* out = (float*)d_out;

    // workspace layout (bytes); total ~151.8 MB (fits the 192 MB L2)
    char* ws = (char*)d_ws;
    size_t off = 0;
    _Float16* xT     = (_Float16*)(ws + off); off += (size_t)NB * HWN * CIN  * 2; // 16.78 MB
    _Float16* wqkv_h = (_Float16*)(ws + off); off += (size_t)CQKV * CIN      * 2;
    _Float16* wproj_h= (_Float16*)(ws + off); off += (size_t)COUT * CATT     * 2;
    float*    alpha  = (float*)   (ws + off); off += COUT * 4;
    float*    biasv  = (float*)   (ws + off); off += COUT * 4;
    _Float16* qkv_h  = (_Float16*)(ws + off); off += (size_t)NB * CQKV * HWN * 2; // 50.33 MB
    _Float16* agg_h  = (_Float16*)(ws + off); off += (size_t)NB * CQKV * HWN * 2; // 50.33 MB
    float*    kvbuf  = (float*)   (ws + off); off += (size_t)NB * NHEAD * 72 * 4;
    _Float16* attT   = (_Float16*)(ws + off); off += (size_t)NB * HWN * CATT * 2; // 33.55 MB
    (void)ws_size; (void)in_sizes; (void)n_in; (void)out_size;

    // P0: weight conversions + BN fold
    k_cvt_f16<<<(CQKV * CIN + 255) / 256, 256, 0, stream>>>(w_qkv, wqkv_h, CQKV * CIN);
    k_cvt_f16<<<(COUT * CATT + 255) / 256, 256, 0, stream>>>(w_proj, wproj_h, COUT * CATT);
    k_bn_fold<<<1, 256, 0, stream>>>(gamma, beta, mean, var, alpha, biasv);

    // P1: transpose x -> xT f16
    k_xpose<<<dim3(HWN / 32, CIN / 32, NB), dim3(32, 8), 0, stream>>>(x, xT);

    // P2: qkv GEMM  (48 Mtiles x 64 Ngroups x 8 batches, 1 wave/block)
    k_gemm_qkv<<<dim3(HWN / 64, CQKV / 16, NB), 32, 0, stream>>>(wqkv_h, xT, qkv_h);

    // P3: depthwise + grouped pointwise -> agg
    k_dwpw<<<dim3(WID / 16, WID / 16, NB * NG), dim3(16, 16), 0, stream>>>(
        qkv_h, w_dw, w_pw, agg_h);

    // P4: kv per (b,head), 4-wave split-K
    k_attn_kv<<<NB * NHEAD, 128, 0, stream>>>(qkv_h, agg_h, kvbuf);

    // P5: q*kv + normalize -> attT
    k_attn_out<<<dim3(HWN / 256, NHEAD, NB), 256, 0, stream>>>(qkv_h, agg_h, kvbuf, attT);

    // P6: proj GEMM + BN -> d_out
    k_gemm_proj<<<dim3(HWN / 64, COUT / 16, NB), 32, 0, stream>>>(
        wproj_h, attT, alpha, biasv, out);
}
